// Retriever_91130616087124
// MI455X (gfx1250) — matrix operand order repaired and checked by reading, not compiled, not run on hardware
//
#include <hip/hip_runtime.h>
#include <hip/hip_bf16.h>
#include <math.h>

// ---------------------------------------------------------------------------
// Retriever scorer for MI455X (gfx1250, wave32, WMMA).
// Hot loop: [E,2068]x[2068,512] GEMM in bf16 WMMA (v_wmma_f32_16x16x32_bf16)
// fused with relu-bias epilogue and the final [512]->1 reduction.
// ---------------------------------------------------------------------------

#define EMBD 512
#define KTOT 2068      // 512 q | 512 x_src | 10 P_src | 512 ea | 512 x_dst | 10 P_dst
#define NKT  65        // ceil(2068/32) K-tiles of 32 (padded with zeros)
#define BM   64        // edges per workgroup
#define APAD 40        // LDS row stride in bf16 (80B -> 20*m mod 64 banks, conflict-free)

typedef __attribute__((ext_vector_type(16))) __bf16          v16bf;
typedef __attribute__((ext_vector_type(8)))  float           v8f;
typedef __attribute__((ext_vector_type(8)))  unsigned short  v8us;

__device__ __forceinline__ unsigned short bf16bits(float f) {
    unsigned u = __float_as_uint(f);
    u += 0x7FFFu + ((u >> 16) & 1u);           // round-to-nearest-even
    return (unsigned short)(u >> 16);
}

// ----------------------------- prep kernels --------------------------------

__global__ void fill_kernel(float* __restrict__ p, int n, float v) {
    int i = blockIdx.x * blockDim.x + threadIdx.x;
    if (i < n) p[i] = v;
}

__global__ void init_out_kernel(float* __restrict__ out, const float* __restrict__ b2, int n) {
    int i = blockIdx.x * blockDim.x + threadIdx.x;
    if (i < n) out[i] = b2[0];
}

__global__ void degree_kernel(const int* __restrict__ ei, float* __restrict__ degIn,
                              float* __restrict__ degOut, int E) {
    int e = blockIdx.x * blockDim.x + threadIdx.x;
    if (e < E) {
        atomicAdd(&degOut[ei[e]], 1.0f);           // count over src
        atomicAdd(&degIn[ei[(long)E + e]], 1.0f);  // count over dst
    }
}

// one wave per node row: is_zero[i] = all(x[i,:]==0)
__global__ void zero_mask_kernel(const float* __restrict__ x, int* __restrict__ isz, int N) {
    int gid  = blockIdx.x * blockDim.x + threadIdx.x;
    int w    = gid >> 5;
    int lane = gid & 31;
    if (w >= N) return;
    const float4* r = (const float4*)(x + (long)w * EMBD);
    float a = 0.0f;
#pragma unroll
    for (int i = 0; i < 4; ++i) {
        float4 v = r[lane + 32 * i];
        a += fabsf(v.x) + fabsf(v.y) + fabsf(v.z) + fabsf(v.w);
    }
    int nz = __any(a != 0.0f);
    if (lane == 0) isz[w] = (nz == 0) ? 1 : 0;
}

__global__ void copy_topic_kernel(const float* __restrict__ topic, float* __restrict__ P, int N) {
    int i = blockIdx.x * blockDim.x + threadIdx.x;
    if (i < N) {
        P[(long)i * 10 + 0] = topic[(long)i * 2 + 0];
        P[(long)i * 10 + 1] = topic[(long)i * 2 + 1];
    }
}

// T[to[e]] += P[from[e], inOff:inOff+2]
__global__ void scatter_kernel(const float* __restrict__ P, int inOff, float* __restrict__ T,
                               const int* __restrict__ from, const int* __restrict__ to, int E) {
    int e = blockIdx.x * blockDim.x + threadIdx.x;
    if (e < E) {
        int f = from[e], t = to[e];
        atomicAdd(&T[(long)t * 2 + 0], P[(long)f * 10 + inOff + 0]);
        atomicAdd(&T[(long)t * 2 + 1], P[(long)f * 10 + inOff + 1]);
    }
}

__global__ void normalize_kernel(float* __restrict__ P, int outOff, const float* __restrict__ T,
                                 const float* __restrict__ deg, int N) {
    int i = blockIdx.x * blockDim.x + threadIdx.x;
    if (i < N) {
        float dg = fmaxf(deg[i], 1.0f);
        P[(long)i * 10 + outOff + 0] = T[(long)i * 2 + 0] / dg;
        P[(long)i * 10 + outOff + 1] = T[(long)i * 2 + 1] / dg;
    }
}

// W1 [2068,512] f32 -> bf16 in B-fragment layout, contiguous K per lane half (same order as loadA):
// block (nt,kt): lane l holds N = nt*16 + (l&15), K = kt*32 + (l<16?0:16)+i, i=0..15
__global__ void swizzle_w1_kernel(const float* __restrict__ W1, unsigned short* __restrict__ Wsw) {
    int idx = blockIdx.x * blockDim.x + threadIdx.x;
    if (idx >= NKT * 32 * 32) return;
    int lane = idx & 31;
    int blk  = idx >> 5;
    int kt   = blk % NKT;
    int nt   = blk / NKT;
    int n    = nt * 16 + (lane & 15);
    int kb   = kt * 32 + ((lane < 16) ? 0 : 16);
    unsigned short* dst = Wsw + ((long)(nt * NKT + kt) * 512 + lane * 16);
#pragma unroll
    for (int i = 0; i < 16; ++i) {
        int k = kb + i;
        float v = (k < KTOT) ? W1[(long)k * EMBD + n] : 0.0f;
        dst[i] = bf16bits(v);
    }
}

// --------------------------- fused GEMM kernel -----------------------------
// 256 threads = 8 waves. Wave w owns N-tiles [4w, 4w+4) (covers all 512 cols),
// and all 4 M-tiles (64 edges). K streamed in 32-wide tiles through a
// double-buffered LDS stage; B fragments ping-pong between two statically
// indexed register sets (K loop unrolled by 2 -> no register copies).

__global__ __launch_bounds__(256) void fused_gemm_kernel(
    const float* __restrict__ x, const int* __restrict__ ei,
    const float* __restrict__ ea, const float* __restrict__ q,
    const float* __restrict__ ne, const float* __restrict__ b1,
    const float* __restrict__ w2, const float* __restrict__ P,
    const int* __restrict__ isz, const unsigned short* __restrict__ Wsw,
    float* __restrict__ out, int E)
{
    __shared__ __align__(16) unsigned short As[2][BM * APAD];

    const int t    = threadIdx.x;
    const int lane = t & 31;
    const int wave = t >> 5;
    const long eBase = (long)blockIdx.x * BM;

    // --- staging setup: each thread owns (row = t/4, 8 K-cols = (t%4)*8) ---
    const int srow = t >> 2;
    const int scol = (t & 3) << 3;
    const long se  = eBase + srow;
    const bool ev  = se < E;
    const long seC = ev ? se : 0;
    const int  s   = ei[seC];
    const int  d   = ei[(long)E + seC];
    const float* qrow  = q  + seC * EMBD;
    const float* earow = ea + seC * EMBD;
    const float* xs = isz[s] ? ne : (x + (long)s * EMBD);
    const float* xd = isz[d] ? ne : (x + (long)d * EMBD);
    const float* Ps = P + (long)s * 10;
    const float* Pd = P + (long)d * 10;

    // per-element fallback (only for runs straddling a segment boundary)
    auto gatherK = [&](int k) -> float {
        if (k < 512)   return qrow[k];
        if (k < 1024)  return xs[k - 512];
        if (k < 1034)  return Ps[k - 1024];
        if (k < 1546)  return earow[k - 1034];
        if (k < 2058)  return xd[k - 1546];
        if (k < KTOT)  return Pd[k - 2058];
        return 0.0f;
    };

    // stage one 8-wide contiguous K-run: resolve the segment ONCE per run,
    // then do 8 straight contiguous loads (fast path for 62/65 K-windows).
    auto stage = [&](int buf, int kt) {
        const int k0 = kt * 32 + scol;
        float f[8];
        if (!ev) {
#pragma unroll
            for (int j = 0; j < 8; ++j) f[j] = 0.0f;
        } else {
            const float* pb = nullptr;
            int lo = 0;
            if (k0 + 8 <= 512)                     { pb = qrow;  lo = k0; }
            else if (k0 >= 512  && k0 + 8 <= 1024) { pb = xs;    lo = k0 - 512; }
            else if (k0 >= 1024 && k0 + 8 <= 1034) { pb = Ps;    lo = k0 - 1024; }
            else if (k0 >= 1034 && k0 + 8 <= 1546) { pb = earow; lo = k0 - 1034; }
            else if (k0 >= 1546 && k0 + 8 <= 2058) { pb = xd;    lo = k0 - 1546; }
            else if (k0 >= 2058 && k0 + 8 <= 2068) { pb = Pd;    lo = k0 - 2058; }
            if (pb) {
#pragma unroll
                for (int j = 0; j < 8; ++j) f[j] = pb[lo + j];
            } else {
#pragma unroll
                for (int j = 0; j < 8; ++j) f[j] = gatherK(k0 + j);
            }
        }
        unsigned u[4];
#pragma unroll
        for (int p = 0; p < 4; ++p)
            u[p] = (unsigned)bf16bits(f[2 * p]) | ((unsigned)bf16bits(f[2 * p + 1]) << 16);
        uint4 uu; uu.x = u[0]; uu.y = u[1]; uu.z = u[2]; uu.w = u[3];
        *(uint4*)&As[buf][srow * APAD + scol] = uu;  // 16B aligned
    };

    // A fragment (contiguous K order, same as Wsw B): lane holds M=lane&15,
    // K = i + (lane<16?0:16)  -> one contiguous 16-elem run.
    auto loadA = [&](int buf, int mt) -> v16bf {
        const int arow = mt * 16 + (lane & 15);
        const int cb   = (lane < 16) ? 0 : 16;
        const unsigned short* base = &As[buf][arow * APAD + cb];
        v8us lo = *(const v8us*)(base);
        v8us hi = *(const v8us*)(base + 8);
        v16bf r;
#pragma unroll
        for (int i = 0; i < 8; ++i) {
            r[i]     = __builtin_bit_cast(__bf16, (unsigned short)lo[i]);
            r[i + 8] = __builtin_bit_cast(__bf16, (unsigned short)hi[i]);
        }
        return r;
    };

    const int ntb = wave * 4;
    auto loadB = [&](int j, int kt) -> v16bf {
        const unsigned short* bp =
            Wsw + ((long)(ntb + j) * NKT + kt) * 512 + lane * 16;
        v8us lo = *(const v8us*)(bp);
        v8us hi = *(const v8us*)(bp + 8);
        v16bf r;
#pragma unroll
        for (int i = 0; i < 8; ++i) {
            r[i]     = __builtin_bit_cast(__bf16, (unsigned short)lo[i]);
            r[i + 8] = __builtin_bit_cast(__bf16, (unsigned short)hi[i]);
        }
        return r;
    };

    v8f acc[4][4];
    const v8f z = {0.f, 0.f, 0.f, 0.f, 0.f, 0.f, 0.f, 0.f};
#pragma unroll
    for (int j = 0; j < 4; ++j)
#pragma unroll
        for (int mt = 0; mt < 4; ++mt) acc[j][mt] = z;

    // two statically-indexed B register sets (ping-pong, no copies)
    v16bf bbuf[2][4];

    // one K-step: compute kt from buffer `cur`, prefetch kt+1 into `cur^1`.
    // Always called with literal cur -> bbuf indices fold to fixed registers.
    auto step = [&](int kt, int cur, bool more) {
        if (more) {
            stage(cur ^ 1, kt + 1);
#pragma unroll
            for (int j = 0; j < 4; ++j) bbuf[cur ^ 1][j] = loadB(j, kt + 1);
        }
        v16bf afrag[4];
#pragma unroll
        for (int mt = 0; mt < 4; ++mt) afrag[mt] = loadA(cur, mt);
#pragma unroll
        for (int j = 0; j < 4; ++j)
#pragma unroll
            for (int mt = 0; mt < 4; ++mt)
                acc[j][mt] = __builtin_amdgcn_wmma_f32_16x16x32_bf16(
                    false, afrag[mt], false, bbuf[cur][j],
                    (short)0, acc[j][mt], false, false);
        __syncthreads();
    };

    // prologue: stage LDS buf0 and prefetch B fragments for kt=0
#pragma unroll
    for (int j = 0; j < 4; ++j) bbuf[0][j] = loadB(j, 0);
    stage(0, 0);
    __syncthreads();

    // NKT = 65 (odd): pairs cover kt = 0..63, tail handles kt = 64 (cur = 0)
    for (int kt = 0; kt < NKT - 1; kt += 2) {
        step(kt, 0, true);
        step(kt + 1, 1, (kt + 2) < NKT);
    }
    step(NKT - 1, 0, false);

    // --- epilogue: relu(acc + b1) * W2, reduce N across lanes, atomic out ---
    float b1v[4], w2v[4];
#pragma unroll
    for (int j = 0; j < 4; ++j) {
        const int n = (ntb + j) * 16 + (lane & 15);
        b1v[j] = b1[n];
        w2v[j] = w2[n];
    }

#pragma unroll
    for (int mt = 0; mt < 4; ++mt) {
#pragma unroll
        for (int r = 0; r < 8; ++r) {
            float v = 0.0f;
#pragma unroll
            for (int j = 0; j < 4; ++j) {
                float h = acc[j][mt][r] + b1v[j];
                v += (h > 0.0f) ? h * w2v[j] : 0.0f;
            }
            // reduce over the 16 lanes of each half (C layout: lane = N)
            v += __shfl_xor(v, 1, 32);
            v += __shfl_xor(v, 2, 32);
            v += __shfl_xor(v, 4, 32);
            v += __shfl_xor(v, 8, 32);
            if ((lane & 15) == 0) {
                // lanes 0..15 hold M = r, lanes 16..31 hold M = r + 8
                const long e = eBase + mt * 16 + r + ((lane >> 4) << 3);
                if (e < E) atomicAdd(&out[e], v);
            }
        }
    }
}

// ------------------------------- launcher ----------------------------------

extern "C" void kernel_launch(void* const* d_in, const int* in_sizes, int n_in,
                              void* d_out, int out_size, void* d_ws, size_t ws_size,
                              hipStream_t stream) {
    (void)n_in; (void)out_size; (void)ws_size;

    const float* x     = (const float*)d_in[0];
    const int*   ei    = (const int*)d_in[1];
    const float* ea    = (const float*)d_in[2];
    const float* topic = (const float*)d_in[3];
    const float* q     = (const float*)d_in[4];
    const float* ne    = (const float*)d_in[5];
    const float* W1    = (const float*)d_in[6];
    const float* b1    = (const float*)d_in[7];
    const float* w2    = (const float*)d_in[8];
    const float* b2    = (const float*)d_in[9];
    float* out = (float*)d_out;

    const int N = in_sizes[0] / EMBD;
    const int E = in_sizes[2] / EMBD;

    // workspace carving (256B aligned)
    char* ws = (char*)d_ws;
    size_t off = 0;
    auto carve = [&](size_t bytes) -> char* {
        char* p = ws + off;
        off = (off + bytes + 255) & ~(size_t)255;
        return p;
    };
    float* P       = (float*)carve((size_t)N * 10 * sizeof(float));
    float* T       = (float*)carve((size_t)N * 2 * sizeof(float));
    float* degIn   = (float*)carve((size_t)N * sizeof(float));
    float* degOut  = (float*)carve((size_t)N * sizeof(float));
    int*   isz     = (int*)carve((size_t)N * sizeof(int));
    unsigned short* Wsw = (unsigned short*)carve((size_t)NKT * 32 * 512 * sizeof(unsigned short));

    const int B  = 256;
    const int gN = (N + B - 1) / B;
    const int gE = (E + B - 1) / B;
    const int g2N = (2 * N + B - 1) / B;
    const int* srcI = ei;
    const int* dstI = ei + E;

    fill_kernel<<<gN, B, 0, stream>>>(degIn, N, 0.0f);
    fill_kernel<<<gN, B, 0, stream>>>(degOut, N, 0.0f);
    degree_kernel<<<gE, B, 0, stream>>>(ei, degIn, degOut, E);
    zero_mask_kernel<<<(N * 32 + B - 1) / B, B, 0, stream>>>(x, isz, N);
    copy_topic_kernel<<<gN, B, 0, stream>>>(topic, P, N);

    // forward round 1: topic (cols 0-1) -> cols 2-3
    fill_kernel<<<g2N, B, 0, stream>>>(T, 2 * N, 0.0f);
    scatter_kernel<<<gE, B, 0, stream>>>(P, 0, T, srcI, dstI, E);
    normalize_kernel<<<gN, B, 0, stream>>>(P, 2, T, degIn, N);
    // forward round 2: cols 2-3 -> cols 4-5
    fill_kernel<<<g2N, B, 0, stream>>>(T, 2 * N, 0.0f);
    scatter_kernel<<<gE, B, 0, stream>>>(P, 2, T, srcI, dstI, E);
    normalize_kernel<<<gN, B, 0, stream>>>(P, 4, T, degIn, N);
    // reverse round 1: topic -> cols 6-7 (edges flipped)
    fill_kernel<<<g2N, B, 0, stream>>>(T, 2 * N, 0.0f);
    scatter_kernel<<<gE, B, 0, stream>>>(P, 0, T, dstI, srcI, E);
    normalize_kernel<<<gN, B, 0, stream>>>(P, 6, T, degOut, N);
    // reverse round 2: cols 6-7 -> cols 8-9
    fill_kernel<<<g2N, B, 0, stream>>>(T, 2 * N, 0.0f);
    scatter_kernel<<<gE, B, 0, stream>>>(P, 6, T, dstI, srcI, E);
    normalize_kernel<<<gN, B, 0, stream>>>(P, 8, T, degOut, N);

    const int nsw = NKT * 32 * 32;
    swizzle_w1_kernel<<<(nsw + B - 1) / B, B, 0, stream>>>(W1, Wsw);

    init_out_kernel<<<gE, B, 0, stream>>>(out, b2, E);

    fused_gemm_kernel<<<(E + BM - 1) / BM, B, 0, stream>>>(
        x, ei, ea, q, ne, b1, w2, P, isz, Wsw, out, E);
}
